// SNN_77756087927356
// MI455X (gfx1250) — matrix operand order repaired and checked by reading, not compiled, not run on hardware
//
#include <hip/hip_runtime.h>

// ---------------- types / constants ----------------
typedef __attribute__((ext_vector_type(16))) __bf16 v16bf;
typedef __attribute__((ext_vector_type(8)))  float  v8f;

#define DTM   0.1f   // DT * TAU_MEM_INV
#define SYND  0.8f   // 1 - DT * TAU_SYN_INV

#define SEQ   128
#define BATCH 256
#define F_IN  2048
#define F_HID 1024
#define F_OUT 20
#define F_OUTP 32    // padded

// workspace layout (bytes)
#define WS_ZIN   0ull                                  // (32768,1024) bf16
#define WS_WIN   67108864ull                           // W_in^T  (2048,1024) bf16 K-major
#define WS_WCI   71303168ull                           // W_ci^T  (1024,1024) bf16 K-major
#define WS_WCR   73400320ull                           // W_cr^T  (1024,1024) bf16 K-major
#define WS_WOUT  75497472ull                           // W_out^T (1024,32)   bf16 K-major
#define WS_V     75563008ull                           // (256,1024) f32
#define WS_I     76611584ull
#define WS_Z0    77660160ull                           // (256,1024) bf16
#define WS_Z1    78184448ull
#define WS_VO    78708736ull                           // (256,20) f32
#define WS_IO    78729216ull
#define WS_PART  78749696ull                           // 4 x (256,1024) f32 split-K partials
#define WS_END   82944000ull
#define WS_STATE_BYTES (WS_END - WS_V)                 // zeroed each launch

#define PART_STRIDE (BATCH * F_HID)                    // elems per partial slice
#define LDB  136                                       // LDS B row stride, BN=128 kernels
#define LDB2 72                                        // LDS B row stride, BN=64 kernels

// ---------------- helpers ----------------
__device__ __forceinline__ unsigned short f2bf(float f) {
  unsigned int u = __float_as_uint(f);
  return (unsigned short)((u + 0x7FFFu + ((u >> 16) & 1u)) >> 16);
}

// A fragment: base = LDS row RM; lane<16: rows M=lane, K {0-7,16-23};
// lane>=16: rows M=lane-16, K {8-15,24-31}. ldr*2 bytes must be 16B aligned.
__device__ __forceinline__ v16bf ldsA_frag(const unsigned short* base, int ldr, int lane) {
  const unsigned short* p = base + (lane & 15) * ldr + ((lane >> 4) << 3);
  v16bf a;
  ((uint4*)&a)[0] = *(const uint4*)(p);
  ((uint4*)&a)[1] = *(const uint4*)(p + 16);
  return a;
}

// B fragment: lane l holds column n0+(l&15); K {0-7,16-23} (l<16) or {8-15,24-31}; LDS K-major, row stride ldr elems
__device__ __forceinline__ v16bf ldsB_frag(const unsigned short* lds, int ldr, int n0, int lane) {
  const unsigned short* p = lds + ((lane >> 4) << 3) * ldr + n0 + (lane & 15);
  v16bf b;
  #pragma unroll
  for (int s = 0; s < 8; ++s) {
    ((unsigned short*)&b)[s]     = p[s * ldr];
    ((unsigned short*)&b)[8 + s] = p[(16 + s) * ldr];
  }
  return b;
}

// ---------------- setup kernels ----------------
// d[(K,N) K-major bf16] = transpose(s[(Nvalid,K) f32]); rows >= Nvalid are zero-padded.
__global__ void __launch_bounds__(256) transp_bf16_k(const float* __restrict__ s,
                                                     unsigned short* __restrict__ d,
                                                     int N, int K, int Nvalid) {
  __shared__ float tile[32][33];
  const int bk = blockIdx.x * 32, bn = blockIdx.y * 32;
  const int tx = threadIdx.x & 31, ty = threadIdx.x >> 5;   // 32 x 8
  #pragma unroll
  for (int j = 0; j < 4; ++j) {
    int n = bn + ty + j * 8;
    tile[ty + j * 8][tx] = (n < Nvalid) ? s[(size_t)n * K + bk + tx] : 0.0f;
  }
  __syncthreads();
  #pragma unroll
  for (int j = 0; j < 4; ++j)
    d[(size_t)(bk + ty + j * 8) * N + bn + tx] = f2bf(tile[tx][ty + j * 8]);
}

__global__ void __launch_bounds__(256) zero_k(uint4* __restrict__ p, int n) {
  int idx = blockIdx.x * 256 + threadIdx.x;
  if (idx < n) { uint4 z; z.x = z.y = z.z = z.w = 0u; p[idx] = z; }
}

// ---------------- z_in = x @ W_in^T  (M=32768,N=1024,K=2048) ----------------
// BM=128 BN=64 BK=32, double-buffered LDS; 8 waves (4x2), wave tile 32x32.
// Wave tile kept small so the whole loop fits in the 128-VGPR budget (no scratch spills).
__global__ void __launch_bounds__(256) zin_gemm_k(const float* __restrict__ x,
                                                  const unsigned short* __restrict__ winT,
                                                  unsigned short* __restrict__ zin) {
  __shared__ unsigned short sA[2][128 * 40];
  __shared__ unsigned short sB[2][32 * LDB2];
  const int tid = threadIdx.x, lane = tid & 31, w = tid >> 5;
  const int m0 = blockIdx.x * 128, n0g = blockIdx.y * 64;
  const int mw = (w & 3) * 32, nw = (w >> 2) * 32;

  const int mA = tid & 127, kA = (tid >> 7) * 16;    // A: 16 floats / thread
  const int kBr = tid >> 3, nB8 = (tid & 7) * 8;     // B: row kBr (0..31), 8 cols
  const float* ga = x + (size_t)(m0 + mA) * F_IN + kA;
  const unsigned short* gb = winT + (size_t)kBr * F_HID + n0g + nB8;

  float4 fa[4];
  uint4  rb;
  auto ldg = [&](int k0) {
    const float* p = ga + k0;
    fa[0] = *(const float4*)(p);
    fa[1] = *(const float4*)(p + 4);
    fa[2] = *(const float4*)(p + 8);
    fa[3] = *(const float4*)(p + 12);
    rb = *(const uint4*)(gb + (size_t)k0 * F_HID);
  };
  auto sts = [&](int b) {
    uint4 o0, o1;
    o0.x = (unsigned)f2bf(fa[0].x) | ((unsigned)f2bf(fa[0].y) << 16);
    o0.y = (unsigned)f2bf(fa[0].z) | ((unsigned)f2bf(fa[0].w) << 16);
    o0.z = (unsigned)f2bf(fa[1].x) | ((unsigned)f2bf(fa[1].y) << 16);
    o0.w = (unsigned)f2bf(fa[1].z) | ((unsigned)f2bf(fa[1].w) << 16);
    o1.x = (unsigned)f2bf(fa[2].x) | ((unsigned)f2bf(fa[2].y) << 16);
    o1.y = (unsigned)f2bf(fa[2].z) | ((unsigned)f2bf(fa[2].w) << 16);
    o1.z = (unsigned)f2bf(fa[3].x) | ((unsigned)f2bf(fa[3].y) << 16);
    o1.w = (unsigned)f2bf(fa[3].z) | ((unsigned)f2bf(fa[3].w) << 16);
    unsigned short* sp = sA[b] + mA * 40 + kA;
    *(uint4*)(sp)     = o0;
    *(uint4*)(sp + 8) = o1;
    *(uint4*)(sB[b] + kBr * LDB2 + nB8) = rb;
  };

  v8f acc[2][2] = {};
  ldg(0);
  sts(0);
  __syncthreads();
  const int NIT = F_IN / 32;
  for (int i = 0; i < NIT; ++i) {
    const int cur = i & 1;
    if (i + 1 < NIT) ldg((i + 1) * 32);          // prefetch next tile (global)
    v16bf af[2], bb[2];
    #pragma unroll
    for (int ii = 0; ii < 2; ++ii) af[ii] = ldsA_frag(sA[cur] + (mw + ii * 16) * 40, 40, lane);
    #pragma unroll
    for (int j = 0; j < 2; ++j) bb[j] = ldsB_frag(sB[cur], LDB2, nw + j * 16, lane);
    #pragma unroll
    for (int ii = 0; ii < 2; ++ii)
      #pragma unroll
      for (int j = 0; j < 2; ++j)
        acc[ii][j] = __builtin_amdgcn_wmma_f32_16x16x32_bf16(
            false, af[ii], false, bb[j], (short)0, acc[ii][j], false, false);
    if (i + 1 < NIT) sts((i + 1) & 1);           // fill other buffer
    __syncthreads();
  }
  const int rl = lane & 15, rh = (lane >> 4) * 8;
  #pragma unroll
  for (int i = 0; i < 2; ++i)
    #pragma unroll
    for (int j = 0; j < 2; ++j)
      #pragma unroll
      for (int r = 0; r < 8; ++r) {
        int gm = m0 + mw + i * 16 + r + rh;
        int gn = n0g + nw + j * 16 + rl;
        zin[(size_t)gm * F_HID + gn] = f2bf(acc[i][j][r]);
      }
}

// -------- split-K recurrent GEMM: partial[kc] = A_kc @ W_kc  (M=256,N=1024,K=512) --------
// grid (4,8,4); kc<2 -> zin_t/WciT, else z_prev/WcrT. Pure overwrite => deterministic.
__global__ void __launch_bounds__(256) rec_gemm_k(const unsigned short* __restrict__ zin,
                                                  const unsigned short* __restrict__ zprev,
                                                  const unsigned short* __restrict__ wciT,
                                                  const unsigned short* __restrict__ wcrT,
                                                  float* __restrict__ part, int t) {
  __shared__ unsigned short sA[2][64 * 40];
  __shared__ unsigned short sB[2][32 * LDB];
  const int tid = threadIdx.x, lane = tid & 31, w = tid >> 5;
  const int m0 = blockIdx.x * 64, n0g = blockIdx.y * 128, kc = blockIdx.z;
  const int mw = (w & 1) * 32, nw = (w >> 1) * 32;

  const unsigned short* Abase;     // row-major (M, F_HID), K-window [kc*512, +512)
  const unsigned short* Wbase;     // K-major (512, 1024) slice
  if (kc < 2) {
    Abase = zin + (size_t)t * BATCH * F_HID + kc * 512;
    Wbase = wciT + (size_t)(kc * 512) * F_HID;
  } else {
    Abase = zprev + (kc - 2) * 512;
    Wbase = wcrT + (size_t)((kc - 2) * 512) * F_HID;
  }
  float* pout = part + (size_t)kc * PART_STRIDE;

  const int mA = tid & 63,  kA = (tid >> 6) * 8;     // A: 8 bf16 / thread
  const int kBr = tid >> 4, nB8 = (tid & 15) * 8;    // B: rows kBr, kBr+16; 8 cols
  const unsigned short* ga = Abase + (size_t)(m0 + mA) * F_HID + kA;
  const unsigned short* gb = Wbase + (size_t)kBr * F_HID + n0g + nB8;

  uint4 ra, rb[2];
  auto ldg = [&](int k0) {
    ra    = *(const uint4*)(ga + k0);
    rb[0] = *(const uint4*)(gb + (size_t)k0 * F_HID);
    rb[1] = *(const uint4*)(gb + (size_t)(k0 + 16) * F_HID);
  };
  auto sts = [&](int b) {
    *(uint4*)(sA[b] + mA * 40 + kA) = ra;
    *(uint4*)(sB[b] + kBr * LDB + nB8)        = rb[0];
    *(uint4*)(sB[b] + (kBr + 16) * LDB + nB8) = rb[1];
  };

  v8f acc[2][2] = {};
  ldg(0);
  sts(0);
  __syncthreads();
  const int NIT = 512 / 32;
  for (int i = 0; i < NIT; ++i) {
    const int cur = i & 1;
    if (i + 1 < NIT) ldg((i + 1) * 32);
    v16bf af[2], bb[2];
    #pragma unroll
    for (int ii = 0; ii < 2; ++ii) af[ii] = ldsA_frag(sA[cur] + (mw + ii * 16) * 40, 40, lane);
    #pragma unroll
    for (int j = 0; j < 2; ++j) bb[j] = ldsB_frag(sB[cur], LDB, nw + j * 16, lane);
    #pragma unroll
    for (int ii = 0; ii < 2; ++ii)
      #pragma unroll
      for (int j = 0; j < 2; ++j)
        acc[ii][j] = __builtin_amdgcn_wmma_f32_16x16x32_bf16(
            false, af[ii], false, bb[j], (short)0, acc[ii][j], false, false);
    if (i + 1 < NIT) sts((i + 1) & 1);
    __syncthreads();
  }
  const int rl = lane & 15, rh = (lane >> 4) * 8;
  #pragma unroll
  for (int i = 0; i < 2; ++i)
    #pragma unroll
    for (int j = 0; j < 2; ++j)
      #pragma unroll
      for (int r = 0; r < 8; ++r) {
        int gm = m0 + mw + i * 16 + r + rh;
        int gn = n0g + nw + j * 16 + rl;
        pout[(size_t)gm * F_HID + gn] = acc[i][j][r];
      }
}

// -------- fused: i-reduction + LIF + output projection + LI update (per step) --------
// 16 blocks of 16 batch rows; 8 waves: nw = w&1 (N half), ks = w>>1: wave K-split 4-way.
// B fragments are read directly from global WoutT (K-major, L2-resident).
__global__ void __launch_bounds__(256) lif_out_k(float* __restrict__ v,
                                                 float* __restrict__ ibuf,
                                                 const float* __restrict__ part,
                                                 unsigned short* __restrict__ zcur,
                                                 const unsigned short* __restrict__ woutT,
                                                 float* __restrict__ vo,
                                                 float* __restrict__ io,
                                                 float* __restrict__ out, int t) {
  __shared__ unsigned short sA[16 * 40];
  __shared__ float sRed[4 * 16 * 32];
  const int tid = threadIdx.x, lane = tid & 31, w = tid >> 5;
  const int m0 = blockIdx.x * 16;
  const int nw = (w & 1) * 16;   // which 16-col half of padded 32 outputs
  const int ks = w >> 1;         // K-split slot 0..3
  v8f acc = {};

  for (int i = 0; i < 32; ++i) {
    const int k0 = i * 32;
    { // fused: i_t = 0.8*i_{t-1} + sum(partials); LIF; stage z into LDS as A operand
      int m_l = tid & 15;
      int kb  = (tid >> 4) * 2;
      size_t base = (size_t)(m0 + m_l) * F_HID + k0 + kb;
      float2 vv   = *(const float2*)(v + base);
      float2 iold = *(const float2*)(ibuf + base);
      float2 p0   = *(const float2*)(part + base);
      float2 p1   = *(const float2*)(part + PART_STRIDE + base);
      float2 p2   = *(const float2*)(part + 2 * PART_STRIDE + base);
      float2 p3   = *(const float2*)(part + 3 * PART_STRIDE + base);
      float in0 = SYND * iold.x + ((p0.x + p1.x) + (p2.x + p3.x));
      float in1 = SYND * iold.y + ((p0.y + p1.y) + (p2.y + p3.y));
      float vd0 = (1.0f - DTM) * vv.x + DTM * in0;
      float vd1 = (1.0f - DTM) * vv.y + DTM * in1;
      bool s0 = vd0 > 1.0f, s1 = vd1 > 1.0f;     // V_TH = 1
      unsigned short zb0 = s0 ? (unsigned short)0x3F80 : (unsigned short)0;
      unsigned short zb1 = s1 ? (unsigned short)0x3F80 : (unsigned short)0;
      float2 vn; vn.x = s0 ? 0.0f : vd0; vn.y = s1 ? 0.0f : vd1;   // V_RESET = 0
      float2 inw; inw.x = in0; inw.y = in1;
      *(float2*)(v + base)    = vn;
      *(float2*)(ibuf + base) = inw;
      unsigned zo = (unsigned)zb0 | ((unsigned)zb1 << 16);
      *(unsigned*)(sA + m_l * 40 + kb) = zo;     // A operand for out-proj
      *(unsigned*)(zcur + base)        = zo;     // z_t for next step's rec GEMM
    }
    __syncthreads();
    if (ks == (i & 3)) {                          // wave-uniform branch; EXEC all-1s inside
      v16bf a = ldsA_frag(sA, 40, lane);
      const unsigned short* p = woutT + (size_t)(k0 + ((lane >> 4) << 3)) * F_OUTP + nw + (lane & 15);
      v16bf b;
      #pragma unroll
      for (int s = 0; s < 8; ++s) {
        ((unsigned short*)&b)[s]     = p[s * F_OUTP];
        ((unsigned short*)&b)[8 + s] = p[(16 + s) * F_OUTP];
      }
      acc = __builtin_amdgcn_wmma_f32_16x16x32_bf16(false, a, false, b, (short)0, acc,
                                                    false, false);
    }
    __syncthreads();
  }

  // cross-wave K reduction through LDS
  const int rl = lane & 15, rh = (lane >> 4) * 8;
  #pragma unroll
  for (int r = 0; r < 8; ++r)
    sRed[(ks * 16 + r + rh) * 32 + nw + rl] = acc[r];
  __syncthreads();

  // final LI update: 16x32 elems, 2 per thread
  #pragma unroll
  for (int c = 0; c < 2; ++c) {
    int e = tid * 2 + c;
    int m_l = e >> 5, n = e & 31;
    float y = (sRed[(m_l) * 32 + n] + sRed[(16 + m_l) * 32 + n]) +
              (sRed[(32 + m_l) * 32 + n] + sRed[(48 + m_l) * 32 + n]);
    if (n < F_OUT) {
      int gm = m0 + m_l;
      size_t idx = (size_t)gm * F_OUT + n;
      float io_old = io[idx];
      float vo_new = (1.0f - DTM) * vo[idx] + DTM * io_old;   // uses io BEFORE adding y
      out[(size_t)t * BATCH * F_OUT + idx] = vo_new;
      vo[idx] = vo_new;
      io[idx] = io_old * SYND + y;
    }
  }
}

// ---------------- host ----------------
extern "C" void kernel_launch(void* const* d_in, const int* in_sizes, int n_in,
                              void* d_out, int out_size, void* d_ws, size_t ws_size,
                              hipStream_t stream) {
  (void)in_sizes; (void)n_in; (void)out_size; (void)ws_size;
  const float* x    = (const float*)d_in[0];
  const float* Win  = (const float*)d_in[1];
  const float* Wci  = (const float*)d_in[2];
  const float* Wcr  = (const float*)d_in[3];
  const float* Wout = (const float*)d_in[4];
  char* ws = (char*)d_ws;
  unsigned short* zin   = (unsigned short*)(ws + WS_ZIN);
  unsigned short* winT  = (unsigned short*)(ws + WS_WIN);
  unsigned short* wciT  = (unsigned short*)(ws + WS_WCI);
  unsigned short* wcrT  = (unsigned short*)(ws + WS_WCR);
  unsigned short* woutT = (unsigned short*)(ws + WS_WOUT);
  float* v    = (float*)(ws + WS_V);
  float* ib   = (float*)(ws + WS_I);
  unsigned short* z0 = (unsigned short*)(ws + WS_Z0);
  unsigned short* z1 = (unsigned short*)(ws + WS_Z1);
  float* vo   = (float*)(ws + WS_VO);
  float* io   = (float*)(ws + WS_IO);
  float* part = (float*)(ws + WS_PART);
  float* out  = (float*)d_out;

  // setup: one-time weight transpose+convert to K-major bf16; zero state
  transp_bf16_k<<<dim3(F_IN / 32,  F_HID / 32), 256, 0, stream>>>(Win,  winT,  F_HID, F_IN,  F_HID);
  transp_bf16_k<<<dim3(F_HID / 32, F_HID / 32), 256, 0, stream>>>(Wci,  wciT,  F_HID, F_HID, F_HID);
  transp_bf16_k<<<dim3(F_HID / 32, F_HID / 32), 256, 0, stream>>>(Wcr,  wcrT,  F_HID, F_HID, F_HID);
  transp_bf16_k<<<dim3(F_HID / 32, 1),          256, 0, stream>>>(Wout, woutT, F_OUTP, F_HID, F_OUT);
  {
    int n16 = (int)(WS_STATE_BYTES / 16);
    zero_k<<<(n16 + 255) / 256, 256, 0, stream>>>((uint4*)(ws + WS_V), n16);
  }

  // big parallel input projection: z_in (bf16)
  zin_gemm_k<<<dim3(SEQ * BATCH / 128, F_HID / 64), 256, 0, stream>>>(x, winT, zin);

  // sequential scan: 2 kernels per step
  unsigned short* zbuf[2] = {z0, z1};
  for (int t = 0; t < SEQ; ++t) {
    lif_out_k<<<BATCH / 16, 256, 0, stream>>>(v, ib, part, zbuf[t & 1], woutT, vo, io, out, t);
    rec_gemm_k<<<dim3(BATCH / 64, F_HID / 128, 4), 256, 0, stream>>>(
        zin, zbuf[(t + 1) & 1], wciT, wcrT, part, t);
  }
}
